// SlotAttention_9277129359459
// MI455X (gfx1250) — matrix operand order repaired and checked by reading, not compiled, not run on hardware
//
#include <hip/hip_runtime.h>
#include <hip/hip_bf16.h>
#include <math.h>

// ---------------------------------------------------------------------------
// Problem constants (from reference)
// ---------------------------------------------------------------------------
static constexpr int BB   = 32;      // batch
static constexpr int NN   = 16384;   // tokens
static constexpr int DIM  = 256;     // input dim
static constexpr int SL   = 8;       // slots
static constexpr int HID  = 128;     // hidden
static constexpr float LN_EPS = 1e-5f;
static constexpr float SCALE  = 0.0625f; // 256^-0.5

typedef _Float16 h16;
typedef __attribute__((ext_vector_type(16))) _Float16 v16h;
typedef __attribute__((ext_vector_type(8)))  float    v8f;

// A-matrix (16x32 f16) K-base for VGPR g, lane-half h (per CDNA5 ISA 7.12.2)
__device__ __forceinline__ int a_kbase(int g, int half) {
    return (g < 4) ? (2 * g + 8 * half) : (16 + 2 * (g - 4) + 8 * half);
}

// ---------------------------------------------------------------------------
// Kernel 0: zero the output (we accumulate with atomics)
// ---------------------------------------------------------------------------
__global__ void k_zero(float* __restrict__ out, int n) {
    int i = blockIdx.x * blockDim.x + threadIdx.x;
    if (i < n) out[i] = 0.0f;
}

// ---------------------------------------------------------------------------
// Kernel 1: prep.  block 0: q = LN(slots) @ Wq^T + bq  -> ws_q (16x128 f16,
// rows 8..15 zero).  blocks 1..128: transpose-convert Wk/Wv (f32 [n][c]) into
// f16 [c][n] so WMMA B fragments are contiguous in N.
// ---------------------------------------------------------------------------
__global__ __launch_bounds__(256)
void k_prep(const float* __restrict__ slots,
            const float* __restrict__ Wq, const float* __restrict__ bq,
            const float* __restrict__ Wk, const float* __restrict__ Wv,
            const float* __restrict__ ln_s_g, const float* __restrict__ ln_s_b,
            h16* __restrict__ ws_q, h16* __restrict__ wsWkT, h16* __restrict__ wsWvT) {
    const int tid = threadIdx.x;
    if (blockIdx.x == 0) {
        __shared__ float sm[SL], sr[SL];
        if (tid < SL) {
            const float* row = slots + tid * HID;
            float s = 0.f, s2 = 0.f;
            for (int c = 0; c < HID; ++c) { float x = row[c]; s += x; s2 += x * x; }
            float m = s / HID;
            float var = s2 / HID - m * m;
            sm[tid] = m;
            sr[tid] = rsqrtf(var + LN_EPS);
        }
        __syncthreads();
        for (int oi = 0; oi < 4; ++oi) {
            int o = oi * 256 + tid;            // 1024 outputs
            int i = o >> 7, h = o & 127;
            const float* row = slots + i * HID;
            float m = sm[i], r = sr[i];
            float acc = bq[h];
            const float* wq = Wq + h * HID;
            for (int c = 0; c < HID; ++c) {
                float xs = (row[c] - m) * r * ln_s_g[c] + ln_s_b[c];
                acc += xs * wq[c];
            }
            ws_q[o] = (h16)acc;
            ws_q[SL * HID + o] = (h16)0.f;     // rows 8..15 -> zero pad
        }
    } else {
        int idx = (blockIdx.x - 1) * 256 + tid;      // 0..32767 over (n,c)
        if (idx < HID * DIM) {
            int n = idx / DIM, c = idx % DIM;
            wsWkT[c * HID + n] = (h16)Wk[idx];
            wsWvT[c * HID + n] = (h16)Wv[idx];
        }
    }
}

// ---------------------------------------------------------------------------
// Kernel 2: fused LayerNorm -> (k,v) WMMA GEMMs -> dots = q @ k^T (WMMA).
// One workgroup = one batch b and a 64-row tile of n.  256 threads = 8 waves.
// Each wave owns one (matrix, M-tile) pair and accumulates all 8 N-tiles in
// registers so the A fragment is loaded from LDS once per K-step.
// ---------------------------------------------------------------------------
__global__ __launch_bounds__(256)
void k_lnkv_dots(const float* __restrict__ inputs,
                 const float* __restrict__ ln_g, const float* __restrict__ ln_b,
                 const float* __restrict__ bk, const float* __restrict__ bv,
                 const h16* __restrict__ wsWkT, const h16* __restrict__ wsWvT,
                 const h16* __restrict__ ws_q,
                 h16* __restrict__ ws_v, float* __restrict__ ws_dots) {
    __shared__ h16 xs[64][DIM];   // LN'd inputs, f16 (32 KB)
    __shared__ h16 ks[64][HID];   // k tile, f16      (16 KB)

    const int b    = blockIdx.x >> 8;        // / 256
    const int tile = blockIdx.x & 255;
    const int n0   = tile * 64;
    const int wave = threadIdx.x >> 5;
    const int lane = threadIdx.x & 31;

    // ---- Phase A: LayerNorm 64 rows, one row per wave-iteration -----------
    for (int r8 = 0; r8 < 8; ++r8) {
        const int row = wave * 8 + r8;
        const float* src = inputs + ((size_t)b * NN + n0 + row) * DIM;
        float vals[8];
        float s = 0.f, s2 = 0.f;
        #pragma unroll
        for (int u = 0; u < 8; ++u) {
            float x = src[lane + 32 * u];
            vals[u] = x; s += x; s2 += x * x;
        }
        #pragma unroll
        for (int m = 16; m >= 1; m >>= 1) {
            s  += __shfl_xor(s,  m, 32);
            s2 += __shfl_xor(s2, m, 32);
        }
        const float mean = s * (1.0f / DIM);
        const float var  = s2 * (1.0f / DIM) - mean * mean;
        const float rstd = rsqrtf(var + LN_EPS);
        #pragma unroll
        for (int u = 0; u < 8; ++u) {
            int c = lane + 32 * u;
            xs[row][c] = (h16)((vals[u] - mean) * rstd * ln_g[c] + ln_b[c]);
        }
    }
    __syncthreads();

    // ---- Phase B: k and v projections via v_wmma_f32_16x16x32_f16 ---------
    // wave 0..3 -> k M-tiles 0..3 (to LDS); wave 4..7 -> v M-tiles 0..3 (to ws)
    const int half = lane >> 4;
    const int mrow = lane & 15;
    {
        const bool isV    = wave >= 4;
        const int  tm     = wave & 3;
        const h16* WT     = isV ? wsWvT : wsWkT;
        const float* bias = isV ? bv : bk;
        const int  am     = tm * 16 + mrow;

        v8f acc[8];
        #pragma unroll
        for (int tn = 0; tn < 8; ++tn) acc[tn] = {};

        for (int kk = 0; kk < 8; ++kk) {
            const int c0 = kk * 32;
            // A fragment (loaded once per K-step): lane=M, VGPR=K pairs
            v16h a;
            #pragma unroll
            for (int g = 0; g < 8; ++g) {
                int kb = c0 + a_kbase(g, half);
                a[2 * g]     = xs[am][kb];
                a[2 * g + 1] = xs[am][kb + 1];
            }
            // B fragments: lane&15 = N, VGPR pair = K (same K order as A)
            const h16* brow = WT + (size_t)(c0 + lane) * HID;
            #pragma unroll
            for (int tn = 0; tn < 8; ++tn) {
                v16h bf;
                #pragma unroll
                for (int g = 0; g < 8; ++g) {
                    int kb = c0 + a_kbase(g, half);
                    bf[2 * g]     = WT[(size_t)kb * HID + tn * 16 + mrow];
                    bf[2 * g + 1] = WT[(size_t)(kb + 1) * HID + tn * 16 + mrow];
                }
                acc[tn] = __builtin_amdgcn_wmma_f32_16x16x32_f16(
                    false, a, false, bf, (short)0, acc[tn], false, false);
            }
        }
        // C layout: VGPR r -> M = r + 8*half, lane&15 -> N
        #pragma unroll
        for (int tn = 0; tn < 8; ++tn) {
            #pragma unroll
            for (int r = 0; r < 8; ++r) {
                const int m = tm * 16 + r + 8 * half;
                const int n = tn * 16 + mrow;
                const h16 outv = (h16)(acc[tn][r] + bias[n]);
                if (isV) ws_v[((size_t)b * NN + n0 + m) * HID + n] = outv;
                else     ks[m][n] = outv;
            }
        }
    }
    __syncthreads();

    // ---- Phase C: dots = q @ k^T  (16x64 output, waves 0..3) --------------
    if (wave < 4) {
        const int tn = wave;                  // local j tile
        v8f acc = {};
        #pragma unroll
        for (int kk = 0; kk < 4; ++kk) {
            const int c0 = kk * 32;           // K over HID
            v16h a;                           // A = padded q (16x128)
            #pragma unroll
            for (int g = 0; g < 8; ++g) {
                int kb = c0 + a_kbase(g, half);
                a[2 * g]     = ws_q[mrow * HID + kb];
                a[2 * g + 1] = ws_q[mrow * HID + kb + 1];
            }
            v16h bf;                          // B[k=h][n=j] = ks[j][h]
            #pragma unroll
            for (int g = 0; g < 8; ++g) {
                int kb = c0 + a_kbase(g, half);
                bf[2 * g]     = ks[tn * 16 + mrow][kb];
                bf[2 * g + 1] = ks[tn * 16 + mrow][kb + 1];
            }
            acc = __builtin_amdgcn_wmma_f32_16x16x32_f16(
                false, a, false, bf, (short)0, acc, false, false);
        }
        if (lane < 16) {                      // only M=0..7 rows are real slots
            const int j = n0 + tn * 16 + mrow;
            #pragma unroll
            for (int r = 0; r < 8; ++r)       // M = r (lanes 0..15)
                ws_dots[((size_t)b * SL + r) * NN + j] = acc[r] * SCALE;
        }
    }
}

// ---------------------------------------------------------------------------
// Kernel 3: per-batch flattened softmax / raw-sum division, 3 Sinkhorn
// iterations, and final slot-normalization, all in one WG per batch.
// ---------------------------------------------------------------------------
__global__ __launch_bounds__(1024)
void k_softmax_sinkhorn(float* __restrict__ dots) {
    __shared__ float red[1024];
    __shared__ float u[SL];
    __shared__ float s_mx, s_rs, s_es;

    const int b   = blockIdx.x;
    const int tid = threadIdx.x;
    float* Q = dots + (size_t)b * SL * NN;
    const int TOT = SL * NN;                 // 131072

    // -- max + raw sum -------------------------------------------------------
    float mx = -3.402823e38f, rs = 0.f;
    for (int i = tid; i < TOT; i += 1024) { float v = Q[i]; mx = fmaxf(mx, v); rs += v; }
    red[tid] = mx; __syncthreads();
    for (int s = 512; s > 0; s >>= 1) { if (tid < s) red[tid] = fmaxf(red[tid], red[tid + s]); __syncthreads(); }
    if (tid == 0) s_mx = red[0];
    __syncthreads();
    red[tid] = rs; __syncthreads();
    for (int s = 512; s > 0; s >>= 1) { if (tid < s) red[tid] += red[tid + s]; __syncthreads(); }
    if (tid == 0) s_rs = red[0];
    __syncthreads();

    // -- exp + sum -----------------------------------------------------------
    const float mxv = s_mx;
    float es = 0.f;
    for (int i = tid; i < TOT; i += 1024) { float e = __expf(Q[i] - mxv); Q[i] = e; es += e; }
    red[tid] = es; __syncthreads();
    for (int s = 512; s > 0; s >>= 1) { if (tid < s) red[tid] += red[tid + s]; __syncthreads(); }
    if (tid == 0) s_es = red[0];
    __syncthreads();
    const float inv = 1.0f / (s_es * s_rs);
    for (int i = tid; i < TOT; i += 1024) Q[i] *= inv;
    __syncthreads();

    // -- 3 Sinkhorn iterations ----------------------------------------------
    const int row = tid >> 7, lanein = tid & 127;
    for (int it = 0; it < 3; ++it) {
        float s = 0.f;
        for (int j = lanein; j < NN; j += 128) s += Q[row * NN + j];
        red[tid] = s; __syncthreads();
        for (int st = 64; st > 0; st >>= 1) { if (lanein < st) red[tid] += red[tid + st]; __syncthreads(); }
        if (lanein == 0) u[row] = red[tid];
        __syncthreads();
        for (int j = tid; j < NN; j += 1024) {
            float vals[SL]; float cs = 0.f;
            #pragma unroll
            for (int i = 0; i < SL; ++i) { vals[i] = Q[i * NN + j]; cs += vals[i]; }
            const float cf = (float)NN / cs;
            #pragma unroll
            for (int i = 0; i < SL; ++i) Q[i * NN + j] = vals[i] * ((float)SL / u[i]) * cf;
        }
        __syncthreads();
    }

    // -- attn = Q / colsum-over-slots ---------------------------------------
    for (int j = tid; j < NN; j += 1024) {
        float vals[SL]; float cs = 0.f;
        #pragma unroll
        for (int i = 0; i < SL; ++i) { vals[i] = Q[i * NN + j]; cs += vals[i]; }
        const float ic = 1.0f / cs;
        #pragma unroll
        for (int i = 0; i < SL; ++i) Q[i * NN + j] = vals[i] * ic;
    }
}

// ---------------------------------------------------------------------------
// Kernel 4: updates = (attn @ v) / SL, split over N with atomic accumulation.
// ---------------------------------------------------------------------------
__global__ __launch_bounds__(256)
void k_updates(const float* __restrict__ attn, const h16* __restrict__ ws_v,
               float* __restrict__ out) {
    __shared__ float at[SL][128];
    const int b     = blockIdx.x >> 3;
    const int chunk = blockIdx.x & 7;
    const int j0    = chunk * 2048;
    const int tid   = threadIdx.x;
    const int h     = tid & 127;
    const int ip    = tid >> 7;              // 0 or 1

    float acc[4] = {0.f, 0.f, 0.f, 0.f};
    for (int blk = 0; blk < 16; ++blk) {
        const int jb = j0 + blk * 128;
        #pragma unroll
        for (int t = 0; t < 4; ++t) {
            int idx = t * 256 + tid;         // 1024 attn values
            int i = idx >> 7, jj = idx & 127;
            at[i][jj] = attn[((size_t)b * SL + i) * NN + jb + jj];
        }
        __syncthreads();
        const h16* vp = ws_v + ((size_t)b * NN + jb) * HID + h;
        if (blk < 15)
            __builtin_prefetch(vp + 128 * HID, 0, 0);   // global_prefetch_b8
        for (int jj = 0; jj < 128; ++jj) {
            const float vv = (float)vp[(size_t)jj * HID];
            #pragma unroll
            for (int s = 0; s < 4; ++s) acc[s] += at[ip + 2 * s][jj] * vv;
        }
        __syncthreads();
    }
    #pragma unroll
    for (int s = 0; s < 4; ++s) {
        const int i = ip + 2 * s;
        atomicAdd(&out[b * (SL * HID) + i * HID + h], acc[s] * (1.0f / SL));
    }
}

// ---------------------------------------------------------------------------
// Launch
// ---------------------------------------------------------------------------
extern "C" void kernel_launch(void* const* d_in, const int* in_sizes, int n_in,
                              void* d_out, int out_size, void* d_ws, size_t ws_size,
                              hipStream_t stream) {
    const float* inputs  = (const float*)d_in[0];
    const float* slots   = (const float*)d_in[1];
    const float* Wq      = (const float*)d_in[2];
    const float* bq      = (const float*)d_in[3];
    const float* Wk      = (const float*)d_in[4];
    const float* bk      = (const float*)d_in[5];
    const float* Wv      = (const float*)d_in[6];
    const float* bv      = (const float*)d_in[7];
    const float* ln_in_g = (const float*)d_in[8];
    const float* ln_in_b = (const float*)d_in[9];
    const float* ln_s_g  = (const float*)d_in[10];
    const float* ln_s_b  = (const float*)d_in[11];
    float* out = (float*)d_out;

    char* ws = (char*)d_ws;
    h16*   ws_q    = (h16*)(ws + 0);                    // 16x128 f16   (4 KB)
    h16*   wsWkT   = (h16*)(ws + 4096);                 // 256x128 f16  (64 KB)
    h16*   wsWvT   = (h16*)(ws + 4096 + 65536);         // 256x128 f16  (64 KB)
    h16*   ws_v    = (h16*)(ws + 262144);               // B*N*H f16    (128 MB)
    float* ws_dots = (float*)(ws + 262144 + (size_t)BB * NN * HID * 2); // 16 MB

    k_zero<<<(BB * SL * HID + 255) / 256, 256, 0, stream>>>(out, BB * SL * HID);
    k_prep<<<1 + (HID * DIM + 255) / 256, 256, 0, stream>>>(
        slots, Wq, bq, Wk, Wv, ln_s_g, ln_s_b, ws_q, wsWkT, wsWvT);
    k_lnkv_dots<<<BB * (NN / 64), 256, 0, stream>>>(
        inputs, ln_in_g, ln_in_b, bk, bv, wsWkT, wsWvT, ws_q, ws_v, ws_dots);
    k_softmax_sinkhorn<<<BB, 1024, 0, stream>>>(ws_dots);
    k_updates<<<BB * 8, 256, 0, stream>>>(ws_dots, ws_v, out);
}
